// ScaledDotProductAttentionWithParaTopicNorm_82197084111116
// MI455X (gfx1250) — matrix operand order repaired and checked by reading, not compiled, not run on hardware
//
#include <hip/hip_runtime.h>

// ---------------------------------------------------------------------------
// ScaledDotProductAttentionWithParaTopicNorm for MI455X (gfx1250, wave32)
// Memory-bound (~1.68 GB traffic); all matmuls via v_wmma_f32_16x16x32_bf16.
// ---------------------------------------------------------------------------

typedef __attribute__((ext_vector_type(16))) __bf16        v16bf;
typedef __attribute__((ext_vector_type(8)))  float         v8f;
typedef __attribute__((ext_vector_type(4)))  unsigned int  u32x4;

union V16 {
  v16bf v;
  u32x4 u[2];
};

#define B_    2
#define H_    16
#define L_    2048
#define DH_   64
#define DM_   1024
#define LOG2E 1.44269504088896340736f

// hardware bf16 convert (v_cvt_pk_bf16_f32)
__device__ __forceinline__ unsigned short f2bf(float f) {
  union { __bf16 b; unsigned short s; } x;
  x.b = (__bf16)f;
  return x.s;
}
__device__ __forceinline__ float bf2f(unsigned short s) {
  union { unsigned int u; float f; } x; x.u = ((unsigned int)s) << 16;
  return x.f;
}

// ------------------------------ prepass ------------------------------------

__global__ void cvt_qk_kernel(const float* __restrict__ q,
                              const float* __restrict__ k,
                              unsigned short* __restrict__ Qb,
                              unsigned short* __restrict__ Kb, int n) {
  int i = blockIdx.x * blockDim.x + threadIdx.x;
  if (i < n) {
    Qb[i] = f2bf(q[i] * 0.125f);   // fold 1/sqrt(64) into Q
    Kb[i] = f2bf(k[i]);
  }
}

// v[B,H,L,DH] f32 -> Vt[B,H,DH,L] bf16 (coalesced writes, l fastest)
__global__ void cvt_v_kernel(const float* __restrict__ v,
                             unsigned short* __restrict__ Vt, int n) {
  int i = blockIdx.x * blockDim.x + threadIdx.x;
  if (i < n) {
    int l    = i & (L_ - 1);
    int rest = i >> 11;          // bh*64 + d
    int d    = rest & 63;
    int bh   = rest >> 6;
    Vt[i] = f2bf(v[((size_t)bh * L_ + l) * DH_ + d]);
  }
}

__global__ void cvt_w_kernel(const float* __restrict__ W,
                             unsigned short* __restrict__ Wb, int n) {
  int i = blockIdx.x * blockDim.x + threadIdx.x;
  if (i < n) Wb[i] = f2bf(W[i]);
}

// --------------------------- attention kernel ------------------------------
// grid: B*H*(L/64) blocks, 128 threads (4 waves); wave = 16 query rows.
// LDS: per-wave 16x2048 bf16 logit cache (64KB) + 16x32 bf16 A-staging (1KB).

__global__ void attn_kernel(const unsigned short* __restrict__ Qb,
                            const unsigned short* __restrict__ Kb,
                            const unsigned short* __restrict__ Vt,
                            const float* __restrict__ bias,
                            const float* __restrict__ pt,
                            float* __restrict__ weights,
                            unsigned short* __restrict__ Ob) {
  extern __shared__ char smem[];
  const int tid  = threadIdx.x;
  const int lane = tid & 31;
  const int wave = tid >> 5;
  const int half = lane >> 4;
  const int ln   = lane & 15;

  const int blk  = blockIdx.x;
  const int bh   = blk >> 5;                  // (L/64) == 32 query blocks
  const int qblk = blk & 31;
  const int q0   = qblk * 64 + wave * 16;

  unsigned short* S  = (unsigned short*)smem + wave * (16 * L_);
  unsigned short* WB = (unsigned short*)(smem + 4 * (16 * L_ * 2)) + wave * (16 * 32);

  // ---- A operands (Q, 16x64 bf16) loaded once: contiguous K layout,
  //      lane m holds K=[0..16), lane m+16 holds K=[16..32)
  V16 aq[2];
  {
    const unsigned short* qrow = Qb + ((size_t)bh * L_ + q0 + ln) * DH_;
#pragma unroll
    for (int s = 0; s < 2; ++s) {
      aq[s].u[0] = *(const u32x4*)(qrow + s * 32 + half * 16);
      aq[s].u[1] = *(const u32x4*)(qrow + s * 32 + half * 16 + 8);
    }
  }

  // ---- pass 1: S = Q*K^T + bias -> LDS (bf16); track row max only.
  // Two INDEPENDENT accumulators (one per 32-wide k-chunk) so the two WMMAs
  // have no D->C dependency: no hazard NOPs, compiler interleaves with loads.
  float mx[8];
#pragma unroll
  for (int r = 0; r < 8; ++r) mx[r] = -3.0e38f;

  const size_t rowBase = (size_t)bh * L_ + q0;    // flat (b,h,q) row index
  for (int kt = 0; kt < L_ / 16; ++kt) {
    if (kt + 1 < L_ / 16)   // gfx1250 global_prefetch_b8 of next K tile
      __builtin_prefetch(Kb + ((size_t)bh * L_ + (kt + 1) * 16 + ln) * DH_, 0, 1);

    V16 bk0, bk1;
    {
      const unsigned short* krow =
          Kb + ((size_t)bh * L_ + kt * 16 + ln) * DH_ + half * 16;
      bk0.u[0] = ((const u32x4*)krow)[0];
      bk0.u[1] = ((const u32x4*)krow)[1];
      bk1.u[0] = ((const u32x4*)(krow + 32))[0];
      bk1.u[1] = ((const u32x4*)(krow + 32))[1];
    }
    v8f c0 = {}, c1 = {};
    c0 = __builtin_amdgcn_wmma_f32_16x16x32_bf16(false, aq[0].v, false, bk0.v,
                                                 (short)0, c0, false, false);
    c1 = __builtin_amdgcn_wmma_f32_16x16x32_bf16(false, aq[1].v, false, bk1.v,
                                                 (short)0, c1, false, false);
#pragma unroll
    for (int r = 0; r < 8; ++r) {
      const int mrow = r + 8 * half;          // C layout: row r+8*(lane/16)
      const size_t idx = (rowBase + mrow) * (size_t)L_ + kt * 16 + ln;
      float vv = (c0[r] + c1[r]) + __builtin_nontemporal_load(&bias[idx]);
      S[mrow * L_ + kt * 16 + ln] = f2bf(vv);
      mx[r] = fmaxf(mx[r], vv * LOG2E);
    }
  }

  // ---- butterfly max across the 16 lanes holding each row (broadcast)
#pragma unroll
  for (int r = 0; r < 8; ++r) {
    float m = mx[r];
#pragma unroll
    for (int off = 1; off < 16; off <<= 1)
      m = fmaxf(m, __shfl_xor(m, off, 32));
    mx[r] = m;
  }

  // ---- vectorized LDS sweep for row sums: lane ln covers cols [ln*128,..)
  float rinv[8];
#pragma unroll
  for (int r = 0; r < 8; ++r) {
    const int mrow = r + 8 * half;
    const unsigned short* srow = S + mrow * L_ + ln * 128;
    float acc = 0.0f;
#pragma unroll 4
    for (int cc = 0; cc < 16; ++cc) {          // 16 x ds_load_b128
      u32x4 pk = ((const u32x4*)srow)[cc];
#pragma unroll
      for (int e = 0; e < 4; ++e) {
        unsigned int w2 = pk[e];
        float lo = bf2f((unsigned short)(w2 & 0xFFFFu));
        float hi = bf2f((unsigned short)(w2 >> 16));
        acc += exp2f(lo * LOG2E - mx[r]) + exp2f(hi * LOG2E - mx[r]);
      }
    }
#pragma unroll
    for (int off = 1; off < 16; off <<= 1)
      acc += __shfl_xor(acc, off, 32);
    rinv[r] = 1.0f / acc;
  }

  // ---- pass 2: w = pt * softmax(S); write weights; O += w @ V (WMMA)
  // o0..o3 are 4 independent accumulator chains: hazard slots stay covered.
  v8f o0 = {}, o1 = {}, o2 = {}, o3 = {};
  for (int kt = 0; kt < L_ / 16; ++kt) {
#pragma unroll
    for (int r = 0; r < 8; ++r) {
      const int mrow = r + 8 * half;
      const size_t idx = (rowBase + mrow) * (size_t)L_ + kt * 16 + ln;
      float sv = bf2f(S[mrow * L_ + kt * 16 + ln]);
      float w  = __builtin_nontemporal_load(&pt[idx]) *
                 exp2f(sv * LOG2E - mx[r]) * rinv[r];
      __builtin_nontemporal_store(w, &weights[idx]);
      WB[mrow * 32 + (kt & 1) * 16 + ln] = f2bf(w);   // stage [m][k] layout
    }
    if (kt & 1) {
      V16 aw;   // rebuild A operand from LDS staging, contiguous k = 16*half + s
      const unsigned short* wrow = WB + ln * 32;
      aw.u[0] = *(const u32x4*)(wrow + half * 16);
      aw.u[1] = *(const u32x4*)(wrow + half * 16 + 8);
      const int k0 = (kt - 1) * 16 + half * 16;
      const unsigned short* vbase = Vt + ((size_t)bh * DH_) * L_ + k0;
      V16 bv;
#define DO_NT(nt, acc)                                                        \
      {                                                                       \
        const unsigned short* vrow = vbase + (size_t)((nt) * 16 + ln) * L_;   \
        bv.u[0] = ((const u32x4*)vrow)[0];                                    \
        bv.u[1] = ((const u32x4*)vrow)[1];                                    \
        acc = __builtin_amdgcn_wmma_f32_16x16x32_bf16(false, aw.v, false,     \
                  bv.v, (short)0, acc, false, false);                         \
      }
      DO_NT(0, o0) DO_NT(1, o1) DO_NT(2, o2) DO_NT(3, o3)
#undef DO_NT
    }
  }

  // ---- write O (bf16) to workspace in [b*L + q, h*64 + d] layout
  const int b = bh >> 4, h = bh & 15;
  unsigned short* obase = Ob + ((size_t)b * L_ + q0) * DM_ + h * DH_;
#pragma unroll
  for (int r = 0; r < 8; ++r) {
    const int mrow = r + 8 * half;
    unsigned short* orow = obase + (size_t)mrow * DM_;
    orow[0  + ln] = f2bf(o0[r]);
    orow[16 + ln] = f2bf(o1[r]);
    orow[32 + ln] = f2bf(o2[r]);
    orow[48 + ln] = f2bf(o3[r]);
  }
}

// --------------------------- output projection -----------------------------
// graph_out[4096,1024] = Ob[4096,1024] @ W_out[1024,1024]^T + b_out
// One 16x16 tile per wave; K=1024 as two interleaved 16-step WMMA chains.

__global__ void proj_kernel(const unsigned short* __restrict__ Ob,
                            const unsigned short* __restrict__ Wb,
                            const float* __restrict__ b_out,
                            float* __restrict__ out) {
  const int tid  = threadIdx.x;
  const int lane = tid & 31;
  const int wave = tid >> 5;
  const int half = lane >> 4;
  const int ln   = lane & 15;
  const int gid  = blockIdx.x * 8 + wave;   // 16384 tiles total
  const int mt   = gid >> 6;                // 256 M-tiles
  const int nt   = gid & 63;                // 64  N-tiles

  const unsigned short* arow = Ob + (size_t)(mt * 16 + ln) * DM_;
  const unsigned short* brow = Wb + (size_t)(nt * 16 + ln) * DM_;  // W row j
  v8f c0 = {}, c1 = {};
  for (int ks = 0; ks < DM_ / 32; ks += 2) {
    V16 a0, b0, a1, b1;
    a0.u[0] = *(const u32x4*)(arow + ks * 32 + half * 16);
    a0.u[1] = *(const u32x4*)(arow + ks * 32 + half * 16 + 8);
    b0.u[0] = ((const u32x4*)(brow + ks * 32 + half * 16))[0];
    b0.u[1] = ((const u32x4*)(brow + ks * 32 + half * 16))[1];
    a1.u[0] = *(const u32x4*)(arow + (ks + 1) * 32 + half * 16);
    a1.u[1] = *(const u32x4*)(arow + (ks + 1) * 32 + half * 16 + 8);
    b1.u[0] = ((const u32x4*)(brow + (ks + 1) * 32 + half * 16))[0];
    b1.u[1] = ((const u32x4*)(brow + (ks + 1) * 32 + half * 16))[1];
    c0 = __builtin_amdgcn_wmma_f32_16x16x32_bf16(false, a0.v, false, b0.v,
                                                 (short)0, c0, false, false);
    c1 = __builtin_amdgcn_wmma_f32_16x16x32_bf16(false, a1.v, false, b1.v,
                                                 (short)0, c1, false, false);
  }
  const float bo = b_out[nt * 16 + ln];
  float* obase = out + (size_t)(mt * 16) * DM_ + nt * 16 + ln;
#pragma unroll
  for (int r = 0; r < 8; ++r)
    __builtin_nontemporal_store((c0[r] + c1[r]) + bo,
                                &obase[(size_t)(r + 8 * half) * DM_]);
}

// ------------------------------- launch ------------------------------------

extern "C" void kernel_launch(void* const* d_in, const int* in_sizes, int n_in,
                              void* d_out, int out_size, void* d_ws, size_t ws_size,
                              hipStream_t stream) {
  const float* q    = (const float*)d_in[0];
  const float* k    = (const float*)d_in[1];
  const float* v    = (const float*)d_in[2];
  const float* pt   = (const float*)d_in[3];
  const float* bias = (const float*)d_in[4];
  const float* W    = (const float*)d_in[5];
  const float* bo   = (const float*)d_in[6];

  float* graph_out = (float*)d_out;                       // [B, L, 1024]
  float* weights   = graph_out + (size_t)B_ * L_ * DM_;   // [B, H, L, L]

  char* ws = (char*)d_ws;
  unsigned short* Qb = (unsigned short*)(ws);                       // 8 MB
  unsigned short* Kb = (unsigned short*)(ws +  8ull * 1024 * 1024); // 8 MB
  unsigned short* Vt = (unsigned short*)(ws + 16ull * 1024 * 1024); // 8 MB
  unsigned short* Ob = (unsigned short*)(ws + 24ull * 1024 * 1024); // 8 MB
  unsigned short* Wb = (unsigned short*)(ws + 32ull * 1024 * 1024); // 2 MB

  const int nqk = B_ * H_ * L_ * DH_;                     // 4,194,304
  cvt_qk_kernel<<<(nqk + 255) / 256, 256, 0, stream>>>(q, k, Qb, Kb, nqk);
  cvt_v_kernel <<<(nqk + 255) / 256, 256, 0, stream>>>(v, Vt, nqk);
  const int nw = DM_ * DM_;
  cvt_w_kernel <<<(nw + 255) / 256, 256, 0, stream>>>(W, Wb, nw);

  const size_t shbytes = 4 * (16 * L_ * 2) + 4 * (16 * 32 * 2); // 266240 B
  attn_kernel<<<B_ * H_ * (L_ / 64), 128, shbytes, stream>>>(
      Qb, Kb, Vt, bias, pt, weights, Ob);

  proj_kernel<<<(256 * 64) / 8, 256, 0, stream>>>(Ob, Wb, bo, graph_out);
}
